// CRF_44074954391622
// MI455X (gfx1250) — matrix operand order repaired and checked by reading, not compiled, not run on hardware
//
#include <hip/hip_runtime.h>

// CRF loss on MI455X (gfx1250).
//   inputs: emit_scores (S,B,L) f32, T (L,L) f32, labels (B,S) i32, start_id i32
//   output: scalar f32 = log_z - gold
//
// Forward algorithm in the log-semiring rewritten as a real matmul:
//   fwd'[b,j] = log( (exp(fwd[b,:]-m_b) @ exp(T))[b,j] ) + m_b + emit[t,b,j]
// exp(T) precomputed once (bf16 hi/lo split, transposed, TDM-staged into LDS),
// per-step (16x128)@(128x128) product on v_wmma_f32_16x16x32_bf16 with 3-term
// hi/lo error compensation. Per-step emit slabs are double-buffered into LDS
// with the Tensor Data Mover (tensor_load_to_lds + s_wait_tensorcnt), so the
// only recurring HBM traffic is fully overlapped with compute.

#define SLEN 512
#define BATCH 256
#define LNUM 128

typedef __attribute__((ext_vector_type(8)))  __bf16 bf16x8;
typedef __attribute__((ext_vector_type(16))) __bf16 v16bf;
typedef __attribute__((ext_vector_type(8)))  float  v8f;
typedef __attribute__((ext_vector_type(4)))  unsigned int u32x4;
typedef __attribute__((ext_vector_type(8)))  int i32x8;
typedef __attribute__((ext_vector_type(4)))  int i32x4;

__device__ __forceinline__ v16bf frag16(const __bf16* p0, const __bf16* p1) {
  bf16x8 a = *(const bf16x8*)p0;   // 16B LDS load
  bf16x8 b = *(const bf16x8*)p1;   // 16B LDS load
  return __builtin_shufflevector(a, b, 0,1,2,3,4,5,6,7,8,9,10,11,12,13,14,15);
}

__device__ __forceinline__ unsigned lds_off(const void* p) {
  // generic LDS pointer: addr[31:0] is the LDS byte offset (aperture mapping)
  return (unsigned)(unsigned long long)(size_t)p;
}

// TDM 2D tile load: global (row-major, stride_elems between rows) -> LDS.
// data_size_code: 1 = 2B elements, 2 = 4B elements. Tracked with TENSORcnt.
__device__ __forceinline__ void tdm_load_2d(unsigned lds_byte_off, const void* gptr,
                                            unsigned rows, unsigned cols,
                                            unsigned stride_elems,
                                            unsigned data_size_code) {
  unsigned long long ga = (unsigned long long)(size_t)gptr;
  u32x4 g0;
  g0[0] = 1u;                                              // count=1 (valid, user)
  g0[1] = lds_byte_off;                                    // lds_addr
  g0[2] = (unsigned)ga;                                    // global_addr[31:0]
  g0[3] = (unsigned)((ga >> 32) & 0x01FFFFFFu) | (2u << 30); // addr[56:32] | type=2
  i32x8 g1;
  g1[0] = (int)(data_size_code << 16);   // wg_mask=0, data_size
  g1[1] = (int)(cols << 16);             // tensor_dim0[15:0]  (abar addr = 0)
  g1[2] = (int)(rows << 16);             // tensor_dim1[15:0]  (dim0 hi16 = 0)
  g1[3] = (int)(cols << 16);             // tile_dim0          (dim1 hi16 = 0)
  g1[4] = (int)rows;                     // tile_dim1 (tile_dim2 = 0)
  g1[5] = (int)stride_elems;             // tensor_dim0_stride[31:0]
  g1[6] = 0;                             // stride hi / dim1_stride lo
  g1[7] = 0;
  i32x4 z4 = {0, 0, 0, 0};               // groups 2/3: 2-D tensor, unused
  i32x8 z8 = {0, 0, 0, 0, 0, 0, 0, 0};
  __builtin_amdgcn_tensor_load_to_lds(g0, g1, z4, z4, z8, 0);
}

// ---------------------------------------------------------------- prep: E = exp(T)
__global__ __launch_bounds__(256) void crf_prep(const float* __restrict__ T,
                                                __bf16* __restrict__ Ehi,
                                                __bf16* __restrict__ Elo) {
  int i = blockIdx.x * 256 + threadIdx.x;
  if (i < LNUM * LNUM) {
    float e = __expf(T[i]);          // values ~[0.6, 1.7]; safe
    __bf16 h = (__bf16)e;
    Ehi[(i % LNUM) * LNUM + i / LNUM] = h;
    Elo[(i % LNUM) * LNUM + i / LNUM] = (__bf16)(e - (float)h);
  }
}

// ---------------------------------------------------------------- forward scan
// grid = 16 blocks (16 batch rows each), 256 threads = 8 waves.
// wave w owns output columns [16w, 16w+16).
__global__ __launch_bounds__(256) void crf_forward(const float* __restrict__ emit,
                                                   const __bf16* __restrict__ Ehi_g,
                                                   const __bf16* __restrict__ Elo_g,
                                                   float* __restrict__ partial_logz) {
  __shared__ __bf16 Ehi[LNUM * LNUM];   // 32 KB, resident all 512 steps
  __shared__ __bf16 Elo[LNUM * LNUM];   // 32 KB
  __shared__ float  fwd[16 * LNUM];     // 8 KB  forward state
  __shared__ __bf16 Phi[16 * LNUM];     // 4 KB  exp(fwd - rowmax), hi part
  __shared__ __bf16 Plo[16 * LNUM];     // 4 KB  lo part
  __shared__ float  embuf[2][16 * LNUM];// 16 KB double-buffered emit slab (TDM)
  __shared__ float  red[16 * 16];       // 1 KB  reduction scratch
  __shared__ float  rowmax[16];

  const int tid  = threadIdx.x;
  const int lane = tid & 31;
  const int wave = tid >> 5;
  const int b0   = blockIdx.x * 16;

  const int n0   = wave * 16;      // N-tile owned by this wave
  const int hgrp = lane >> 4;      // lane half (A-fragment K grouping)
  const int mrow = lane & 15;      // A-fragment row / C-fragment column

  // ---- TDM-stage E (2x 32KB bf16) and the first two emit slabs -----------
  if (wave == 0) {
    tdm_load_2d(lds_off(Ehi), Ehi_g, LNUM, LNUM, LNUM, 1);
    tdm_load_2d(lds_off(Elo), Elo_g, LNUM, LNUM, LNUM, 1);
    tdm_load_2d(lds_off(&embuf[0][0]), emit + (size_t)b0 * LNUM, 16, LNUM, LNUM, 2);
    tdm_load_2d(lds_off(&embuf[1][0]),
                emit + ((size_t)1 * BATCH + b0) * LNUM, 16, LNUM, LNUM, 2);
    // in-order TDM: <=1 outstanding means E + emit[0] have landed
    __builtin_amdgcn_s_wait_tensorcnt(1);
  }
  __syncthreads();

  // fwd_0 = emit[0]  (reference: no START transition at t=0)
  for (int i = tid; i < 16 * LNUM; i += 256) fwd[i] = embuf[0][i];
  __syncthreads();

  for (int t = 1; t < SLEN; ++t) {
    // ---- TDM pipeline: issue slab t+1, wait for slab t ------------------
    if (wave == 0) {
      if (t + 1 < SLEN) {
        tdm_load_2d(lds_off(&embuf[(t + 1) & 1][0]),
                    emit + ((size_t)(t + 1) * BATCH + b0) * LNUM, 16, LNUM, LNUM, 2);
        __builtin_amdgcn_s_wait_tensorcnt(1);   // slab t landed (in-order TDM)
      } else {
        __builtin_amdgcn_s_wait_tensorcnt(0);   // pipeline drain at last step
      }
    }

    // ---- (a) per-row max of fwd -----------------------------------------
    {
      const int r = tid >> 4, c = tid & 15;
      float mx = -3.0e38f;
      for (int j = c; j < LNUM; j += 16) mx = fmaxf(mx, fwd[r * LNUM + j]);
      red[r * 16 + c] = mx;
    }
    __syncthreads();
    if (tid < 16) {
      float mx = red[tid * 16];
      for (int c = 1; c < 16; ++c) mx = fmaxf(mx, red[tid * 16 + c]);
      rowmax[tid] = mx;
    }
    __syncthreads();
    // ---- (b) P = exp(fwd - rowmax), bf16 hi/lo split --------------------
    {
      const int r = tid >> 4, c = tid & 15;
      const float mx = rowmax[r];
      for (int j = c; j < LNUM; j += 16) {
        float p = __expf(fwd[r * LNUM + j] - mx);   // in (0, 1]
        __bf16 h = (__bf16)p;
        Phi[r * LNUM + j] = h;
        Plo[r * LNUM + j] = (__bf16)(p - (float)h);
      }
    }
    __syncthreads();   // publishes P *and* the TDM-landed emit slab t

    // ---- emit[t] slab now in LDS (TDM); read this wave's tile -----------
    float ev[8];
#pragma unroll
    for (int r = 0; r < 8; ++r) {
      const int M = r + 8 * hgrp;
      ev[r] = embuf[t & 1][M * LNUM + n0 + mrow];
    }

    // ---- C = P @ E via bf16 WMMA, 3-term hi/lo compensation -------------
    v8f acc = 0.0f;
#pragma unroll
    for (int kc = 0; kc < 4; ++kc) {
      // A fragment (16x32, MxK): lane holds K = {8h..8h+7, 16+8h..+7} of chunk
      const int ka = kc * 32 + 8 * hgrp;
      v16bf Ah = frag16(Phi + mrow * LNUM + ka, Phi + mrow * LNUM + ka + 16);
      v16bf Al = frag16(Plo + mrow * LNUM + ka, Plo + mrow * LNUM + ka + 16);
      // B fragment (32x16, KxN): lane holds column N = n0+mrow, K = {8h..8h+7, 16+8h..+7} of chunk (E stored transposed)
      const int kb = (n0 + mrow) * LNUM + ka;
      v16bf Bh = frag16(Ehi + kb, Ehi + kb + 16);
      v16bf Bl = frag16(Elo + kb, Elo + kb + 16);
      acc = __builtin_amdgcn_wmma_f32_16x16x32_bf16(false, Ah, false, Bh, (short)0, acc, false, false);
      acc = __builtin_amdgcn_wmma_f32_16x16x32_bf16(false, Ah, false, Bl, (short)0, acc, false, false);
      acc = __builtin_amdgcn_wmma_f32_16x16x32_bf16(false, Al, false, Bh, (short)0, acc, false, false);
    }

    // ---- fwd' = log(acc) + rowmax + emit[t] -----------------------------
    // C layout: VGPR r, lane: M = r + 8*(lane>>4), N = lane&15
#pragma unroll
    for (int r = 0; r < 8; ++r) {
      const int M = r + 8 * hgrp;
      fwd[M * LNUM + n0 + mrow] = __logf(acc[r]) + rowmax[M] + ev[r];
    }
    __syncthreads();
  }

  // ---- final: log_z partial = sum_b LSE_j fwd[b,j] ----------------------
  {
    const int r = tid >> 4, c = tid & 15;
    float mx = -3.0e38f;
    for (int j = c; j < LNUM; j += 16) mx = fmaxf(mx, fwd[r * LNUM + j]);
    red[r * 16 + c] = mx;
  }
  __syncthreads();
  if (tid < 16) {
    float mx = red[tid * 16];
    for (int c = 1; c < 16; ++c) mx = fmaxf(mx, red[tid * 16 + c]);
    rowmax[tid] = mx;
  }
  __syncthreads();
  {
    const int r = tid >> 4, c = tid & 15;
    const float mx = rowmax[r];
    float s = 0.0f;
    for (int j = c; j < LNUM; j += 16) s += __expf(fwd[r * LNUM + j] - mx);
    red[r * 16 + c] = s;
  }
  __syncthreads();
  if (tid < 16) {
    float s = 0.0f;
    for (int c = 0; c < 16; ++c) s += red[tid * 16 + c];
    rowmax[tid] += __logf(s);          // reuse rowmax[] as per-row LSE
  }
  __syncthreads();
  if (tid == 0) {
    float z = 0.0f;
    for (int r = 0; r < 16; ++r) z += rowmax[r];
    partial_logz[blockIdx.x] = z;      // fixed-order, deterministic
  }
}

// ---------------------------------------------------------------- gold score
// grid = 256 (one block per batch row); deterministic tree reduction.
__global__ __launch_bounds__(256) void crf_gold(const float* __restrict__ emit,
                                                const float* __restrict__ T,
                                                const int* __restrict__ labels,
                                                const int* __restrict__ start_id_p,
                                                float* __restrict__ pgold) {
  const int b = blockIdx.x;
  const int sid = *start_id_p;
  float s = 0.0f;
  for (int t = threadIdx.x; t < SLEN; t += 256) {
    const int cur  = labels[b * SLEN + t];
    const int prev = (t == 0) ? sid : labels[b * SLEN + t - 1];
    s += emit[((size_t)t * BATCH + b) * LNUM + cur] + T[prev * LNUM + cur];
  }
  __shared__ float buf[256];
  buf[threadIdx.x] = s;
  __syncthreads();
  for (int off = 128; off > 0; off >>= 1) {
    if (threadIdx.x < off) buf[threadIdx.x] += buf[threadIdx.x + off];
    __syncthreads();
  }
  if (threadIdx.x == 0) pgold[b] = buf[0];
}

// ---------------------------------------------------------------- combine
__global__ void crf_combine(const float* __restrict__ plogz,
                            const float* __restrict__ pgold,
                            float* __restrict__ out) {
  if (threadIdx.x == 0 && blockIdx.x == 0) {
    float z = 0.0f;
    for (int k = 0; k < 16; ++k) z += plogz[k];
    float g = 0.0f;
    for (int b = 0; b < BATCH; ++b) g += pgold[b];
    out[0] = z - g;
  }
}

extern "C" void kernel_launch(void* const* d_in, const int* in_sizes, int n_in,
                              void* d_out, int out_size, void* d_ws, size_t ws_size,
                              hipStream_t stream) {
  const float* emit   = (const float*)d_in[0];   // (S,B,L) f32
  const float* T      = (const float*)d_in[1];   // (L,L)   f32
  const int*   labels = (const int*)d_in[2];     // (B,S)   i32
  const int*   sid    = (const int*)d_in[3];     // scalar  i32

  char* ws = (char*)d_ws;
  __bf16* Ehi  = (__bf16*)(ws);                  // 32 KB
  __bf16* Elo  = (__bf16*)(ws + 32768);          // 32 KB
  float* plogz = (float*)(ws + 65536);           // 16 floats
  float* pgold = (float*)(ws + 65536 + 64);      // 256 floats
  float* out   = (float*)d_out;

  crf_prep   <<<64,  256, 0, stream>>>(T, Ehi, Elo);
  crf_forward<<<16,  256, 0, stream>>>(emit, Ehi, Elo, plogz);
  crf_gold   <<<256, 256, 0, stream>>>(emit, T, labels, sid, pgold);
  crf_combine<<<1,   32,  0, stream>>>(plogz, pgold, out);
}
